// ODETransformerBlock_20315195310357
// MI455X (gfx1250) — matrix operand order repaired and checked by reading, not compiled, not run on hardware
//
#include <hip/hip_runtime.h>

typedef __attribute__((ext_vector_type(16))) __bf16 v16bf;
typedef __attribute__((ext_vector_type(8)))  __bf16 v8bf;
typedef __attribute__((ext_vector_type(8)))  float  v8f;
typedef __attribute__((ext_vector_type(4)))  unsigned int v4u;
typedef __attribute__((ext_vector_type(8)))  int v8i;
typedef __attribute__((ext_vector_type(4)))  int v4i;

#define ODE_DT 0.01f

// ---------------- WMMA helpers (CDNA5 bf16 16x16x32, f32 accum) -------------
__device__ __forceinline__ v8f wmma_bf16(v16bf a, v16bf b, v8f c) {
  return __builtin_amdgcn_wmma_f32_16x16x32_bf16(false, a, false, b, (short)0, c,
                                                 false, false);
}

// A-matrix 16x32 bf16 fragment from row-major source (row stride ld elements).
// ISA layout: lane l: m=l&15, half=l>>4; elems 0..7 -> k=half*8+0..7,
// elems 8..15 -> k=16+half*8+0..7.
__device__ __forceinline__ v16bf load_a_frag_bf(const __bf16* base, int ld,
                                                int m0, int k0, int lane) {
  int m = m0 + (lane & 15), half = lane >> 4;
  const __bf16* p = base + (size_t)m * ld + k0 + half * 8;
  v8bf lo = *(const v8bf*)p;
  v8bf hi = *(const v8bf*)(p + 16);
  v16bf a;
#pragma unroll
  for (int i = 0; i < 8; ++i) { a[i] = lo[i]; a[8 + i] = hi[i]; }
  return a;
}

// Same A layout but sourced from f32 (e.g. LDS activations), converted to bf16.
__device__ __forceinline__ v16bf load_a_frag_f32(const float* base, int ld,
                                                 int m0, int k0, int lane) {
  int m = m0 + (lane & 15), half = lane >> 4;
  const float* p = base + (size_t)m * ld + k0 + half * 8;
  v16bf a;
#pragma unroll
  for (int i = 0; i < 8; ++i) {
    a[i]     = (__bf16)p[i];
    a[8 + i] = (__bf16)p[i + 16];
  }
  return a;
}

// B-matrix 32x16 bf16 fragment from B^T stored row-major [N][K] (stride ld).
// ISA layout: lane l: n=l&15, half=l>>4; elems 0..7 -> k=half*8+0..7,
// elems 8..15 -> k=16+half*8+0..7.
__device__ __forceinline__ v16bf load_b_frag(const __bf16* bt, int ld,
                                             int n0, int k0, int lane) {
  int n = n0 + (lane & 15), half = lane >> 4;
  const __bf16* p = bt + (size_t)n * ld + k0 + half * 8;
  v8bf lo = *(const v8bf*)p;
  v8bf hi = *(const v8bf*)(p + 16);
  v16bf b;
#pragma unroll
  for (int i = 0; i < 8; ++i) { b[i] = lo[i]; b[8 + i] = hi[i]; }
  return b;
}

// Workgroup LayerNorm over 16 rows x 256 cols; TPR = threads per row.
template <int TPR>
__device__ __forceinline__ void wg_layernorm(const float* buf, float* out,
                                             const float* g, const float* beta,
                                             float* redS, float* redQ, float* mr,
                                             int tid) {
  constexpr int CPT = 256 / TPR;  // cols per thread
  int rowr = tid / TPR, part = tid % TPR;
  const float* rp = buf + rowr * 256 + part * CPT;
  float s = 0.f, s2 = 0.f;
#pragma unroll
  for (int j = 0; j < CPT; ++j) { float v = rp[j]; s += v; s2 += v * v; }
  redS[rowr * TPR + part] = s;
  redQ[rowr * TPR + part] = s2;
  __syncthreads();
  if (part == 0) {
    float ts = 0.f, tq = 0.f;
#pragma unroll
    for (int p = 0; p < TPR; ++p) { ts += redS[rowr * TPR + p]; tq += redQ[rowr * TPR + p]; }
    float m = ts * (1.f / 256.f);
    float var = tq * (1.f / 256.f) - m * m;
    mr[rowr * 2 + 0] = m;
    mr[rowr * 2 + 1] = rsqrtf(var + 1e-5f);
  }
  __syncthreads();
  float m = mr[rowr * 2 + 0], rstd = mr[rowr * 2 + 1];
  float* op = out + rowr * 256 + part * CPT;
#pragma unroll
  for (int j = 0; j < CPT; ++j) {
    int col = part * CPT + j;
    op[j] = (rp[j] - m) * rstd * g[col] + beta[col];
  }
  __syncthreads();
}

// ---------------- TDM: DMA a 256x256 bf16 panel global -> LDS ---------------
__device__ __forceinline__ void wait_tensorcnt0() {
#if __has_builtin(__builtin_amdgcn_s_wait_tensorcnt)
  __builtin_amdgcn_s_wait_tensorcnt(0);
#else
  asm volatile("s_wait_tensorcnt 0x0" ::: "memory");
#endif
}

#if __has_builtin(__builtin_amdgcn_tensor_load_to_lds)
#define HAVE_TDM 1
__device__ __forceinline__ void tdm_load_weight_256x256(const __bf16* gsrc,
                                                        __bf16* ldst) {
  unsigned lds_off = (unsigned)(unsigned long long)(void*)ldst;  // LDS addr = low 32b
  unsigned long long ga = (unsigned long long)gsrc;
  // D# group0: count=1 | lds_addr | global_addr[56:0] | type=2
  v4u g0;
  g0[0] = 1u;
  g0[1] = lds_off;
  g0[2] = (unsigned)(ga & 0xFFFFFFFFu);
  g0[3] = (unsigned)((ga >> 32) & 0x1FFFFFFu) | (2u << 30);
  // D# group1: wg_mask=0, data_size=1(2B), tensor=256x256, tile=256x256,
  // dim0_stride=256 (all in data_size units)
  v8i g1;
  g1[0] = (int)(1u << 16);          // data_size=1 at bits[17:16]
  g1[1] = (int)(0x0100u << 16);     // tensor_dim0[15:0]=256 at bits[63:48]
  g1[2] = (int)(0x0100u << 16);     // tensor_dim1[15:0]=256 at bits[95:80]
  g1[3] = (int)(0x0100u << 16);     // tile_dim0=256 at bits[127:112]
  g1[4] = 0x0100;                   // tile_dim1=256 at bits[143:128]; tile_dim2=0
  g1[5] = 256;                      // tensor_dim0_stride[31:0]=256
  g1[6] = 0;                        // stride hi / dim1_stride lo
  g1[7] = 0;
  v4i z4 = {0, 0, 0, 0};
#if defined(__clang_major__) && (__clang_major__ >= 23)
  v8i z8 = {0, 0, 0, 0, 0, 0, 0, 0};
  __builtin_amdgcn_tensor_load_to_lds(g0, g1, z4, z4, z8, 0);
#else
  __builtin_amdgcn_tensor_load_to_lds(g0, g1, z4, z4, 0);
#endif
}
#else
#define HAVE_TDM 0
#endif

// ---------------- Kernel 1: convert x to bf16, transpose weights to bf16 ----
__global__ void k_convert(const float* __restrict__ x,
                          const float* __restrict__ wqkv,
                          const float* __restrict__ wo,
                          const float* __restrict__ wl1,
                          const float* __restrict__ wl2,
                          __bf16* xb, __bf16* wqkvT, __bf16* woT,
                          __bf16* wl1T, __bf16* wl2T) {
  int i = blockIdx.x * blockDim.x + threadIdx.x;
  if (i < 2048 * 256) xb[i] = (__bf16)x[i];
  if (i < 768 * 256) {               // wqkvT[n][k] = wqkv[k][n]; wqkv is [256][768]
    int n = i >> 8, k = i & 255;
    wqkvT[i] = (__bf16)wqkv[k * 768 + n];
  }
  if (i < 256 * 256) {
    int n = i >> 8, k = i & 255;
    woT[i]  = (__bf16)wo [k * 256 + n];
    wl1T[i] = (__bf16)wl1[k * 256 + n];
    wl2T[i] = (__bf16)wl2[k * 256 + n];
  }
}

// ---------------- Kernel 2: QKV GEMM  (2048x256)@(256x768)+b -----------------
__global__ void k_qkv(const __bf16* __restrict__ xb,
                      const __bf16* __restrict__ wqkvT,
                      const float* __restrict__ bqkv,
                      __bf16* __restrict__ qkvb, __bf16* __restrict__ vT) {
  int lane = threadIdx.x & 31, wave = threadIdx.x >> 5;
  int m0 = blockIdx.x * 16;
  int ncol0 = blockIdx.y * 256 + wave * 64;
  v8f acc[4] = {};
  for (int k0 = 0; k0 < 256; k0 += 32) {
    v16bf a = load_a_frag_bf(xb, 256, m0, k0, lane);
#pragma unroll
    for (int t = 0; t < 4; ++t) {
      v16bf b = load_b_frag(wqkvT, 256, ncol0 + t * 16, k0, lane);
      acc[t] = wmma_bf16(a, b, acc[t]);
    }
  }
  int n_ = lane & 15, mb = (lane >> 4) * 8;
#pragma unroll
  for (int t = 0; t < 4; ++t) {
    int col = ncol0 + t * 16 + n_;
    float bias = bqkv[col];
#pragma unroll
    for (int r = 0; r < 8; ++r) {
      int row = m0 + mb + r;
      float v = acc[t][r] + bias;
      qkvb[row * 768 + col] = (__bf16)v;
      if (col >= 512) {  // V: also store transposed per (b,h): vT[b][h][d][key]
        int d = col - 512, h = d >> 5, dd = d & 31;
        int bi = row >> 8, key = row & 255;
        vT[(((bi * 8 + h) * 32) + dd) * 256 + key] = (__bf16)v;
      }
    }
  }
}

// ---------------- Kernel 3: attention per (b,h) -----------------------------
__global__ void k_attn(const __bf16* __restrict__ qkvb,
                       const __bf16* __restrict__ vT,
                       __bf16* __restrict__ pob) {
  extern __shared__ float sc_mem[];
  int lane = threadIdx.x & 31, wave = threadIdx.x >> 5;
  int bi = blockIdx.x >> 3, h = blockIdx.x & 7;
  float* sc = sc_mem + wave * 16 * 256;
  const float scale = 0.17677669529663689f;  // 1/sqrt(32)
  const __bf16* qbase = qkvb + (size_t)(bi * 256) * 768 + h * 32;
  const __bf16* kbase = qkvb + (size_t)(bi * 256) * 768 + 256 + h * 32;
  const __bf16* vbase = vT + (size_t)((bi * 8 + h) * 32) * 256;
  int n_ = lane & 15, mb = (lane >> 4) * 8;
  for (int it = 0; it < 4; ++it) {
    int q0 = (wave + it * 4) * 16;
    v16bf aq = load_a_frag_bf(qbase, 768, q0, 0, lane);
#pragma unroll
    for (int kt = 0; kt < 16; ++kt) {
      v16bf bk = load_b_frag(kbase, 768, kt * 16, 0, lane);
      v8f accs = {};
      accs = wmma_bf16(aq, bk, accs);
#pragma unroll
      for (int r = 0; r < 8; ++r)
        sc[(mb + r) * 256 + kt * 16 + n_] = accs[r] * scale;
    }
    __syncthreads();
    if (lane < 16) {  // row softmax
      float* row = sc + lane * 256;
      float mx = -1e30f;
      for (int j = 0; j < 256; ++j) mx = fmaxf(mx, row[j]);
      float s = 0.f;
      for (int j = 0; j < 256; ++j) { float e = expf(row[j] - mx); row[j] = e; s += e; }
      float inv = 1.f / s;
      for (int j = 0; j < 256; ++j) row[j] *= inv;
    }
    __syncthreads();
#pragma unroll
    for (int nt = 0; nt < 2; ++nt) {
      v8f acc = {};
      for (int kt = 0; kt < 8; ++kt) {
        v16bf af = load_a_frag_f32(sc, 256, 0, kt * 32, lane);
        v16bf bv = load_b_frag(vbase, 256, nt * 16, kt * 32, lane);
        acc = wmma_bf16(af, bv, acc);
      }
#pragma unroll
      for (int r = 0; r < 8; ++r) {
        int row = bi * 256 + q0 + mb + r;
        int col = h * 32 + nt * 16 + n_;
        pob[row * 256 + col] = (__bf16)acc[r];
      }
    }
    __syncthreads();
  }
}

// ---------------- Kernel 4: out proj + residual + LN1 -> x1 -----------------
__global__ void k_proj_ln(const __bf16* __restrict__ pob,
                          const __bf16* __restrict__ woT,
                          const float* __restrict__ bo,
                          const float* __restrict__ x,
                          const float* __restrict__ g1,
                          const float* __restrict__ b1,
                          float* __restrict__ x1) {
  __shared__ float xs[16 * 256];
  __shared__ float redS[16 * 8], redQ[16 * 8], mr[16 * 2];
  int lane = threadIdx.x & 31, wave = threadIdx.x >> 5;
  int m0 = blockIdx.x * 16;
  int ncol0 = wave * 64;
  v8f acc[4] = {};
  for (int k0 = 0; k0 < 256; k0 += 32) {
    v16bf a = load_a_frag_bf(pob, 256, m0, k0, lane);
#pragma unroll
    for (int t = 0; t < 4; ++t) {
      v16bf b = load_b_frag(woT, 256, ncol0 + t * 16, k0, lane);
      acc[t] = wmma_bf16(a, b, acc[t]);
    }
  }
  int n_ = lane & 15, mb = (lane >> 4) * 8;
#pragma unroll
  for (int t = 0; t < 4; ++t) {
    int col = ncol0 + t * 16 + n_;
#pragma unroll
    for (int r = 0; r < 8; ++r) {
      int row = m0 + mb + r;
      xs[(mb + r) * 256 + col] = acc[t][r] + bo[col] + x[(size_t)row * 256 + col];
    }
  }
  __syncthreads();
  wg_layernorm<8>(xs, x1 + (size_t)m0 * 256, g1, b1, redS, redQ, mr, threadIdx.x);
}

// ---------------- Kernel 5: fused ODE (256 thr = 8 waves) + final LN --------
// GEMM1 B from LDS (TDM-preloaded wl1T); GEMM2 B streamed from L2 (wl2T).
__device__ __forceinline__ void ode_f(const float* argb, float* dst, __bf16* hid,
                                      const __bf16* wl1_lds, const __bf16* wl2T,
                                      const float* bl1, const float* bl2,
                                      const float* gn, const float* bn,
                                      float* redS, float* redQ, float* mr,
                                      int tid) {
  int lane = tid & 31, wave = tid >> 5;   // 8 waves, 32 cols each
  int ncol0 = wave * 32;
  int n_ = lane & 15, mb = (lane >> 4) * 8;
  {  // hidden = relu(arg @ wl1 + bl1), stored bf16 in LDS
    v8f acc[2] = {};
    for (int k0 = 0; k0 < 256; k0 += 32) {
      v16bf a = load_a_frag_f32(argb, 256, 0, k0, lane);
#pragma unroll
      for (int t = 0; t < 2; ++t) {
        v16bf b = load_b_frag(wl1_lds, 256, ncol0 + t * 16, k0, lane);
        acc[t] = wmma_bf16(a, b, acc[t]);
      }
    }
#pragma unroll
    for (int t = 0; t < 2; ++t) {
      int col = ncol0 + t * 16 + n_;
      float bias = bl1[col];
#pragma unroll
      for (int r = 0; r < 8; ++r)
        hid[(mb + r) * 256 + col] = (__bf16)fmaxf(acc[t][r] + bias, 0.f);
    }
  }
  __syncthreads();
  {  // dst = hidden @ wl2 + bl2 + arg  (pre-LN)
    v8f acc[2] = {};
    for (int k0 = 0; k0 < 256; k0 += 32) {
      v16bf a = load_a_frag_bf(hid, 256, 0, k0, lane);
#pragma unroll
      for (int t = 0; t < 2; ++t) {
        v16bf b = load_b_frag(wl2T, 256, ncol0 + t * 16, k0, lane);
        acc[t] = wmma_bf16(a, b, acc[t]);
      }
    }
#pragma unroll
    for (int t = 0; t < 2; ++t) {
      int col = ncol0 + t * 16 + n_;
      float bias = bl2[col];
#pragma unroll
      for (int r = 0; r < 8; ++r)
        dst[(mb + r) * 256 + col] = acc[t][r] + bias + argb[(mb + r) * 256 + col];
    }
  }
  __syncthreads();
  wg_layernorm<16>(dst, dst, gn, bn, redS, redQ, mr, tid);  // LN in place
}

__global__ void k_ode(const float* __restrict__ x1,
                      const float* __restrict__ g2, const float* __restrict__ b2,
                      const __bf16* __restrict__ wl1T, const __bf16* __restrict__ wl2T,
                      const float* __restrict__ bl1, const float* __restrict__ bl2,
                      const float* __restrict__ gn, const float* __restrict__ bn,
                      const int* __restrict__ lead, float* __restrict__ out) {
  extern __shared__ char smem[];
  __bf16* wlds = (__bf16*)smem;                    // 128KB: wl1T panel
  float* y    = (float*)(smem + 131072);
  float* arg  = y + 4096;
  float* k1   = arg + 4096;
  float* k2   = k1 + 4096;
  float* k3   = k2 + 4096;
  float* redS = k3 + 4096;                         // 16*16
  float* redQ = redS + 256;
  float* mr   = redQ + 256;                        // 16*2
  __bf16* hid = (__bf16*)(mr + 32);                // 8KB

  int tid = threadIdx.x;
  int m0 = blockIdx.x * 16;
  int bi = m0 >> 8;
  int steps = lead[bi];

  // Preload wl1T into LDS (TDM async DMA when available).
#if HAVE_TDM
  if (tid < 32) {
    tdm_load_weight_256x256(wl1T, wlds);
    wait_tensorcnt0();
  }
#else
  for (int i = tid * 8; i < 256 * 256; i += 256 * 8)
    *(v8bf*)(wlds + i) = *(const v8bf*)(wl1T + i);
#endif
  for (int i = tid; i < 4096; i += 256) y[i] = x1[(size_t)m0 * 256 + i];
  __syncthreads();

  for (int s = 0; s < steps; ++s) {
    ode_f(y, k1, hid, wlds, wl2T, bl1, bl2, gn, bn, redS, redQ, mr, tid);
    for (int i = tid; i < 4096; i += 256)
      arg[i] = y[i] + (ODE_DT * (1.f / 3.f)) * k1[i];
    __syncthreads();
    ode_f(arg, k2, hid, wlds, wl2T, bl1, bl2, gn, bn, redS, redQ, mr, tid);
    for (int i = tid; i < 4096; i += 256)
      arg[i] = y[i] + ODE_DT * (k2[i] - (1.f / 3.f) * k1[i]);
    __syncthreads();
    ode_f(arg, k3, hid, wlds, wl2T, bl1, bl2, gn, bn, redS, redQ, mr, tid);
    for (int i = tid; i < 4096; i += 256) {
      arg[i] = y[i] + ODE_DT * (k1[i] - k2[i] + k3[i]);
      k1[i]  = k1[i] + 3.f * (k2[i] + k3[i]);      // fold partial RK4 combo
    }
    __syncthreads();
    ode_f(arg, k2, hid, wlds, wl2T, bl1, bl2, gn, bn, redS, redQ, mr, tid);  // k4
    for (int i = tid; i < 4096; i += 256)
      y[i] = y[i] + (ODE_DT * 0.125f) * (k1[i] + k2[i]);
    __syncthreads();
  }

  for (int i = tid; i < 4096; i += 256) arg[i] = x1[(size_t)m0 * 256 + i] + y[i];
  __syncthreads();
  wg_layernorm<16>(arg, out + (size_t)m0 * 256, g2, b2, redS, redQ, mr, tid);
}

// ---------------- host launcher ---------------------------------------------
extern "C" void kernel_launch(void* const* d_in, const int* in_sizes, int n_in,
                              void* d_out, int out_size, void* d_ws, size_t ws_size,
                              hipStream_t stream) {
  const float* x    = (const float*)d_in[0];
  const float* wqkv = (const float*)d_in[1];
  const float* bqkv = (const float*)d_in[2];
  const float* wo   = (const float*)d_in[3];
  const float* bo   = (const float*)d_in[4];
  const float* g1   = (const float*)d_in[5];
  const float* b1   = (const float*)d_in[6];
  const float* g2   = (const float*)d_in[7];
  const float* b2   = (const float*)d_in[8];
  const float* wl1  = (const float*)d_in[9];
  const float* bl1  = (const float*)d_in[10];
  const float* wl2  = (const float*)d_in[11];
  const float* bl2  = (const float*)d_in[12];
  const float* gn   = (const float*)d_in[13];
  const float* bn   = (const float*)d_in[14];
  const int*   lead = (const int*)d_in[15];
  float* out = (float*)d_out;

  char* ws = (char*)d_ws;
  size_t off = 0;
  auto carve = [&](size_t bytes) {
    void* p = ws + off;
    off = (off + bytes + 255) & ~(size_t)255;
    return p;
  };
  __bf16* xb     = (__bf16*)carve(2048 * 256 * 2);
  __bf16* wqkvT  = (__bf16*)carve(768 * 256 * 2);
  __bf16* woT    = (__bf16*)carve(256 * 256 * 2);
  __bf16* wl1T   = (__bf16*)carve(256 * 256 * 2);
  __bf16* wl2T   = (__bf16*)carve(256 * 256 * 2);
  __bf16* qkvb   = (__bf16*)carve((size_t)2048 * 768 * 2);
  __bf16* vT     = (__bf16*)carve(64 * 32 * 256 * 2);
  __bf16* pob    = (__bf16*)carve(2048 * 256 * 2);
  float*  x1     = (float*)carve((size_t)2048 * 256 * 4);

  k_convert<<<2048, 256, 0, stream>>>(x, wqkv, wo, wl1, wl2,
                                      xb, wqkvT, woT, wl1T, wl2T);
  k_qkv<<<dim3(128, 3), 128, 0, stream>>>(xb, wqkvT, bqkv, qkvb, vT);
  k_attn<<<64, 128, 4 * 16 * 256 * sizeof(float), stream>>>(qkvb, vT, pob);
  k_proj_ln<<<128, 128, 0, stream>>>(pob, woT, bo, x, g1, b1, x1);
  size_t ode_smem = 131072                      // wl1T panel in LDS
                  + (size_t)(5 * 4096) * 4      // y, arg, k1, k2, k3
                  + (256 + 256 + 32) * 4        // LN reductions
                  + 4096 * 2;                   // hidden (bf16)
  k_ode<<<128, 256, ode_smem, stream>>>(x1, g2, b2, wl1T, wl2T, bl1, bl2,
                                        gn, bn, lead, out);
  (void)in_sizes; (void)n_in; (void)out_size; (void)ws_size;
}
